// SelfAttention_38508676775944
// MI455X (gfx1250) — matrix operand order repaired and checked by reading, not compiled, not run on hardware
//
#include <hip/hip_runtime.h>
#include <math.h>

// Problem constants (reference: B=8, C=512, T=4096, DK=64, SCALE=8, GAMMA=1)
#define B_     8
#define C_     512
#define T_     4096
#define DK_    64
#define NCOLS  640          // 64 (q) + 64 (k) + 512 (v) packed per token row
#define INV_SCALE 0.125f
#define GAMMA_ 1.0f

typedef _Float16 half_t;
typedef __attribute__((ext_vector_type(4)))  _Float16 v4h;
typedef __attribute__((ext_vector_type(8)))  _Float16 v8h;
typedef __attribute__((ext_vector_type(16))) _Float16 v16h;
typedef __attribute__((ext_vector_type(8)))  float    v8f;
typedef __attribute__((ext_vector_type(4)))  int      v4i;

// Async global->LDS copy (CDNA5 GLOBAL_LOAD_ASYNC_TO_LDS_B128, ASYNCcnt) if the
// toolchain exposes it; otherwise fall back to a b128 register round-trip.
#if defined(__has_builtin)
#if __has_builtin(__builtin_amdgcn_global_load_async_to_lds_b128) && \
    __has_builtin(__builtin_amdgcn_s_wait_asynccnt)
#define HAVE_ASYNC_LDS 1
#endif
#endif
#define AS1 __attribute__((address_space(1)))
#define AS3 __attribute__((address_space(3)))

__device__ __forceinline__ v8f wmma16(v16h a, v16h b, v8f c) {
  // D = A(16x32 f16) * B(32x16 f16) + C(16x16 f32)
  return __builtin_amdgcn_wmma_f32_16x16x32_f16(false, a, false, b, (short)0, c,
                                                false, false);
}

__device__ __forceinline__ v16h cat8(v8h lo, v8h hi) {
  return __builtin_shufflevector(lo, hi, 0, 1, 2, 3, 4, 5, 6, 7,
                                 8, 9, 10, 11, 12, 13, 14, 15);
}

// A fragment (16x32, M x K) from row-major [M][K] LDS tile; ld multiple of 8
// halves so both v8h loads are 16B-aligned ds_load_b128.
__device__ __forceinline__ v16h load_A_frag(const half_t* tile, int ld, int koff) {
  const int lane = threadIdx.x & 31;
  const int m  = lane & 15;
  const int h16 = (lane >> 4) * 16;
  const half_t* row = tile + m * ld + koff + h16;
  const v8h lo = *(const v8h*)(row);
  const v8h hi = *(const v8h*)(row + 8);
  return cat8(lo, hi);
}

// B fragment (32x16, K x N) from a *fragment-major* [N][K] LDS tile:
// lane n holds col N=n with K = h16..h16+15 contiguous -> two ds_load_b128.
__device__ __forceinline__ v16h load_B_fragT(const half_t* tile, int ld, int noff,
                                             int koff) {
  const int lane = threadIdx.x & 31;
  const int n   = lane & 15;
  const int h16 = (lane >> 4) * 16;
  const half_t* row = tile + (noff + n) * ld + koff + h16;
  const v8h lo = *(const v8h*)(row);
  const v8h hi = *(const v8h*)(row + 8);
  return cat8(lo, hi);
}

// ===================== kernel 0: spectral norm (1 block) =====================
__global__ __launch_bounds__(512) void sigma_kernel(const float* __restrict__ Wv,
                                                    const float* __restrict__ u,
                                                    float* __restrict__ inv_sigma_out) {
  __shared__ float vbuf[C_];
  __shared__ float red[C_];
  const int tid = threadIdx.x;
  float acc = 0.0f;                               // v0 = Wv^T u
  for (int o = 0; o < C_; ++o) acc += Wv[o * C_ + tid] * u[o];
  vbuf[tid] = acc;
  red[tid]  = acc * acc;
  __syncthreads();
  for (int st = 256; st > 0; st >>= 1) {
    if (tid < st) red[tid] += red[tid + st];
    __syncthreads();
  }
  const float nrm = sqrtf(red[0]) + 1e-12f;       // ||Wv^T u|| + eps
  __syncthreads();
  float acc2 = 0.0f;                              // u2_0 = Wv (vbuf/nrm)
  for (int c = 0; c < C_; ++c) acc2 += Wv[tid * C_ + c] * vbuf[c];
  acc2 /= nrm;
  red[tid] = acc2 * acc2;
  __syncthreads();
  for (int st = 256; st > 0; st >>= 1) {
    if (tid < st) red[tid] += red[tid + st];
    __syncthreads();
  }
  if (tid == 0) {
    const float n2sq  = red[0];                        // ||Wv v||^2
    const float sigma = n2sq / (sqrtf(n2sq) + 1e-12f); // u2 . (Wv v)
    *inv_sigma_out = 1.0f / sigma;
  }
}

// ========== kernel 1: fused q/k/v projection  Y = X^T @ [Wq^T|Wk^T|Wv_sn^T] ==========
#define ASLD 40    // 32 + pad (80B rows, 16B aligned)
#define BTLD 40    // Bt is fragment-major [640][32] + pad

__global__ __launch_bounds__(256) void proj_kernel(const float* __restrict__ x,
                                                   const float* __restrict__ Wq,
                                                   const float* __restrict__ Wk,
                                                   const float* __restrict__ Wv,
                                                   const float* __restrict__ inv_sigma_p,
                                                   half_t* __restrict__ y) {
  __shared__ half_t As[16 * ASLD];
  __shared__ half_t Bt[NCOLS * BTLD];
  const int tid  = threadIdx.x;
  const int wid  = tid >> 5;
  const int lane = tid & 31;
  const int t0   = blockIdx.x * 16;
  const int b    = blockIdx.y;
  const float inv_sigma = *inv_sigma_p;
  const size_t xb = (size_t)b * C_ * T_;

  v8f acc[5];
#pragma unroll
  for (int i = 0; i < 5; ++i)
    acc[i] = (v8f){0.f, 0.f, 0.f, 0.f, 0.f, 0.f, 0.f, 0.f};

  for (int k0 = 0; k0 < C_; k0 += 32) {
    __syncthreads();
    // stage A = X^T[t0:t0+16, k0:k0+32]: float4 along t (contiguous in x)
    if (tid < 128) {
      const int k  = tid & 31;
      const int m4 = (tid >> 5) * 4;
      const float4 w = *(const float4*)&x[xb + (size_t)(k0 + k) * T_ + (t0 + m4)];
      As[(m4 + 0) * ASLD + k] = (half_t)w.x;
      As[(m4 + 1) * ASLD + k] = (half_t)w.y;
      As[(m4 + 2) * ASLD + k] = (half_t)w.z;
      As[(m4 + 3) * ASLD + k] = (half_t)w.w;
    }
    // stage Bt[n][k] = [Wq^T|Wk^T|(Wv/sigma)^T] fragment-major; weight rows are
    // contiguous along k -> float4 global loads + v4h (b64) LDS stores
    for (int g = tid; g < NCOLS * 8; g += 256) {
      const int n  = g >> 3;
      const int k4 = (g & 7) * 4;
      const float* src;
      float s;
      if (n < 64)       { src = Wq + n * C_;         s = 1.0f; }
      else if (n < 128) { src = Wk + (n - 64) * C_;  s = 1.0f; }
      else              { src = Wv + (n - 128) * C_; s = inv_sigma; }
      const float4 w = *(const float4*)(src + k0 + k4);
      v4h h;
      h[0] = (half_t)(w.x * s); h[1] = (half_t)(w.y * s);
      h[2] = (half_t)(w.z * s); h[3] = (half_t)(w.w * s);
      *(v4h*)&Bt[n * BTLD + k4] = h;
    }
    __syncthreads();
    // preload all fragments, then issue the WMMA chain back-to-back
    const v16h a = load_A_frag(As, ASLD, 0);
    v16h bf[5];
#pragma unroll
    for (int tt = 0; tt < 5; ++tt)
      bf[tt] = load_B_fragT(Bt, BTLD, wid * 80 + tt * 16, 0);
#pragma unroll
    for (int tt = 0; tt < 5; ++tt)
      acc[tt] = wmma16(a, bf[tt], acc[tt]);
  }
  // store f16 rows: y[b, t, 0:64]=q, [64:128]=k, [128:640]=v
  const int hf = lane >> 4;
  const int n  = lane & 15;
#pragma unroll
  for (int tt = 0; tt < 5; ++tt) {
    const int col = wid * 80 + tt * 16 + n;
#pragma unroll
    for (int r = 0; r < 8; ++r) {
      const int m = r + 8 * hf;
      y[(size_t)(b * T_ + t0 + m) * NCOLS + col] = (half_t)acc[tt][r];
    }
  }
}

// ========== kernel 2: flash attention  out = softmax(QK^T/8) V  (+bv, +x) ==========
#define QLD  72    // 64 + pad (144B rows)
#define KLD  72
#define VTLD 40    // Vt is fragment-major [512][32] + pad
#define PLD  40    // 32 + pad

__global__ __launch_bounds__(256) void attn_kernel(const half_t* __restrict__ y,
                                                   const float* __restrict__ x,
                                                   const float* __restrict__ bvec,
                                                   float* __restrict__ out) {
  __shared__ half_t Qs[16 * QLD];
  __shared__ half_t Ks[32 * KLD];
  __shared__ half_t Vt[512 * VTLD];
  __shared__ half_t Ps[16 * PLD];
  __shared__ float  Es[16 * 32];
  __shared__ float  m_sm[16], l_sm[16], scale_sm[16];

  const int tid  = threadIdx.x;
  const int wid  = tid >> 5;
  const int lane = tid & 31;
  const int t0   = blockIdx.x * 16;
  const int b    = blockIdx.y;
  const size_t ybase = (size_t)b * T_ * NCOLS;

  // load the 16 query rows once (v8h = b128 copies)
  if (tid < 128) {
    const int m  = tid >> 3;
    const int d8 = (tid & 7) * 8;
    *(v8h*)&Qs[m * QLD + d8] = *(const v8h*)&y[ybase + (size_t)(t0 + m) * NCOLS + d8];
  }
  if (tid < 16) { m_sm[tid] = -INFINITY; l_sm[tid] = 0.0f; }
  __syncthreads();

  const v16h aq0 = load_A_frag(Qs, QLD, 0);
  const v16h aq1 = load_A_frag(Qs, QLD, 32);

  v8f oacc[4];
#pragma unroll
  for (int i = 0; i < 4; ++i)
    oacc[i] = (v8f){0.f, 0.f, 0.f, 0.f, 0.f, 0.f, 0.f, 0.f};

  for (int s0 = 0; s0 < T_; s0 += 32) {
    __syncthreads();  // previous iteration finished reading Ks/Vt/Ps
    // prefetch next K/V row block into cache while this block computes
    if (lane == 0 && s0 + 32 < T_)
      __builtin_prefetch(&y[ybase + (size_t)(s0 + 32 + wid * 4) * NCOLS + 64], 0, 0);
    // stage K block [32 x 64]: async global->LDS b128 when available
    {
      const int s  = tid >> 3;
      const int d8 = (tid & 7) * 8;
      const half_t* gsrc = &y[ybase + (size_t)(s0 + s) * NCOLS + 64 + d8];
      half_t* ldst = &Ks[s * KLD + d8];
#if defined(HAVE_ASYNC_LDS)
      __builtin_amdgcn_global_load_async_to_lds_b128((AS1 v4i*)gsrc,
                                                     (AS3 v4i*)ldst, 0, 0);
#else
      *(v8h*)ldst = *(const v8h*)gsrc;
#endif
    }
    // stage V block transposed -> Vt[c][s] (fragment-major for GEMM2):
    // 4x4 register transpose, v4h loads/stores
    for (int g = tid; g < 1024; g += 256) {
      const int s4 = (g & 7) * 4;
      const int c4 = (g >> 3) * 4;
      const size_t yrow = ybase + (size_t)(s0 + s4) * NCOLS + 128 + c4;
      const v4h r0 = *(const v4h*)&y[yrow];
      const v4h r1 = *(const v4h*)&y[yrow + NCOLS];
      const v4h r2 = *(const v4h*)&y[yrow + 2 * NCOLS];
      const v4h r3 = *(const v4h*)&y[yrow + 3 * NCOLS];
      const v4h c0 = {r0[0], r1[0], r2[0], r3[0]};
      const v4h c1 = {r0[1], r1[1], r2[1], r3[1]};
      const v4h c2 = {r0[2], r1[2], r2[2], r3[2]};
      const v4h c3 = {r0[3], r1[3], r2[3], r3[3]};
      *(v4h*)&Vt[(c4 + 0) * VTLD + s4] = c0;
      *(v4h*)&Vt[(c4 + 1) * VTLD + s4] = c1;
      *(v4h*)&Vt[(c4 + 2) * VTLD + s4] = c2;
      *(v4h*)&Vt[(c4 + 3) * VTLD + s4] = c3;
    }
#if defined(HAVE_ASYNC_LDS)
    __builtin_amdgcn_s_wait_asynccnt(0);
#endif
    __syncthreads();

    // waves 0 and 1: each computes one 16x16 tile of E = Q K^T / 8
    if (wid < 2) {
      const int nt = wid;
      const v16h bk0 = load_B_fragT(Ks, KLD, nt * 16, 0);
      const v16h bk1 = load_B_fragT(Ks, KLD, nt * 16, 32);
      v8f e_acc = (v8f){0.f, 0.f, 0.f, 0.f, 0.f, 0.f, 0.f, 0.f};
      e_acc = wmma16(aq0, bk0, e_acc);
      e_acc = wmma16(aq1, bk1, e_acc);
      const int hf = lane >> 4;
      const int n  = lane & 15;
#pragma unroll
      for (int r = 0; r < 8; ++r)
        Es[(r + 8 * hf) * 32 + nt * 16 + n] = e_acc[r] * INV_SCALE;
    }
    __syncthreads();

    // online softmax: all 32 lanes of wave 0; each half-wave reduces 16 cols,
    // cross-half combine via __shfl_xor(16) (wave32 native)
    if (wid == 0) {
      const int row = lane & 15;
      const int hf  = lane >> 4;
      const float mold = m_sm[row];
      const float* erow = &Es[row * 32 + hf * 16];
      float pmax = -INFINITY;
#pragma unroll
      for (int j = 0; j < 16; ++j) pmax = fmaxf(pmax, erow[j]);
      const float omax = __shfl_xor(pmax, 16, 32);
      const float mx = fmaxf(fmaxf(pmax, omax), mold);
      float rs = 0.0f;
#pragma unroll
      for (int j4 = 0; j4 < 4; ++j4) {
        v4h pp;
#pragma unroll
        for (int j = 0; j < 4; ++j) {
          const float p = __expf(erow[j4 * 4 + j] - mx);
          rs += p;
          pp[j] = (half_t)p;
        }
        *(v4h*)&Ps[row * PLD + hf * 16 + j4 * 4] = pp;
      }
      rs += __shfl_xor(rs, 16, 32);
      if (hf == 0) {
        const float sc = __expf(mold - mx);
        m_sm[row]     = mx;
        l_sm[row]     = l_sm[row] * sc + rs;
        scale_sm[row] = sc;
      }
    }
    __syncthreads();

    // all 8 waves: rescale running O, preload 4 B fragments, then 4 WMMAs
    float scr[8];
#pragma unroll
    for (int r = 0; r < 8; ++r) scr[r] = scale_sm[r + 8 * (lane >> 4)];
#pragma unroll
    for (int ct = 0; ct < 4; ++ct)
#pragma unroll
      for (int r = 0; r < 8; ++r) oacc[ct][r] *= scr[r];

    const v16h ap = load_A_frag(Ps, PLD, 0);
    v16h bvf[4];
#pragma unroll
    for (int ct = 0; ct < 4; ++ct)
      bvf[ct] = load_B_fragT(Vt, VTLD, wid * 64 + ct * 16, 0);
#pragma unroll
    for (int ct = 0; ct < 4; ++ct)
      oacc[ct] = wmma16(ap, bvf[ct], oacc[ct]);
  }
  __syncthreads();

  // epilogue: normalize, add bias (sum attn == 1 folds bv here), residual, store.
  const int hf = lane >> 4;
  float linv[8];
#pragma unroll
  for (int r = 0; r < 8; ++r) linv[r] = 1.0f / l_sm[r + 8 * hf];
  const int n = lane & 15;
#pragma unroll
  for (int ct = 0; ct < 4; ++ct) {
    const int cch    = wid * 64 + ct * 16 + n;
    const float bias = bvec[cch];
    const size_t base = ((size_t)b * C_ + cch) * T_ + t0 + 8 * hf;
    const float4 x0 = *(const float4*)&x[base];
    const float4 x1 = *(const float4*)&x[base + 4];
    float4 o0, o1;
    o0.x = GAMMA_ * (oacc[ct][0] * linv[0] + bias) + x0.x;
    o0.y = GAMMA_ * (oacc[ct][1] * linv[1] + bias) + x0.y;
    o0.z = GAMMA_ * (oacc[ct][2] * linv[2] + bias) + x0.z;
    o0.w = GAMMA_ * (oacc[ct][3] * linv[3] + bias) + x0.w;
    o1.x = GAMMA_ * (oacc[ct][4] * linv[4] + bias) + x1.x;
    o1.y = GAMMA_ * (oacc[ct][5] * linv[5] + bias) + x1.y;
    o1.z = GAMMA_ * (oacc[ct][6] * linv[6] + bias) + x1.z;
    o1.w = GAMMA_ * (oacc[ct][7] * linv[7] + bias) + x1.w;
    *(float4*)&out[base]     = o0;
    *(float4*)&out[base + 4] = o1;
  }
}

extern "C" void kernel_launch(void* const* d_in, const int* in_sizes, int n_in,
                              void* d_out, int out_size, void* d_ws, size_t ws_size,
                              hipStream_t stream) {
  const float* x  = (const float*)d_in[0];
  const float* Wq = (const float*)d_in[1];
  const float* Wk = (const float*)d_in[2];
  const float* Wv = (const float*)d_in[3];
  const float* bv = (const float*)d_in[4];
  const float* u  = (const float*)d_in[5];
  float* out = (float*)d_out;

  float*  inv_sigma = (float*)d_ws;                 // 1 float
  half_t* y = (half_t*)((char*)d_ws + 256);         // [B, T, 640] f16 (~42 MB)

  sigma_kernel<<<dim3(1), dim3(512), 0, stream>>>(Wv, u, inv_sigma);
  proj_kernel<<<dim3(T_ / 16, B_), dim3(256), 0, stream>>>(x, Wq, Wk, Wv, inv_sigma, y);
  attn_kernel<<<dim3(T_ / 16, B_), dim3(256), 0, stream>>>(y, x, bv, out);
}
